// _LegacyMultiheadAttention_72713796321899
// MI455X (gfx1250) — matrix operand order repaired and checked by reading, not compiled, not run on hardware
//
#include <hip/hip_runtime.h>
#include <hip/hip_bf16.h>
#include <math.h>

// ---------------------------------------------------------------------------
// Problem constants
// ---------------------------------------------------------------------------
#define Tn 1024
#define Bn 4
#define En 1024
#define Hn 16
#define Dn 64
#define Mn (Tn*Bn)   // 4096 rows for the big GEMMs

typedef __attribute__((ext_vector_type(16))) __bf16 v16bf;
typedef __attribute__((ext_vector_type(8)))  float  v8f;
typedef __attribute__((ext_vector_type(4)))  unsigned int u32x4;
typedef __attribute__((ext_vector_type(8)))  unsigned int u32x8;

union FragAB { uint4 u[2]; v16bf v; };   // 32 bytes = 16 bf16 per lane

// ---------------------------------------------------------------------------
// fp32 -> bf16 conversion (grid-stride)
// ---------------------------------------------------------------------------
__global__ void cvt_bf16_kernel(const float* __restrict__ in,
                                __bf16* __restrict__ out, int n) {
  for (int i = blockIdx.x * blockDim.x + threadIdx.x; i < n;
       i += gridDim.x * blockDim.x)
    out[i] = (__bf16)in[i];
}

// ---------------------------------------------------------------------------
// Relative-position bias table: bias depends only on delta = s - t.
// out[h][delta+1023], index 2047 unused (zeroed).
// ---------------------------------------------------------------------------
__global__ void bias_table_kernel(const float* __restrict__ tbl,
                                  float* __restrict__ out) {
  const int h = blockIdx.x;
  for (int i = threadIdx.x; i < 2048; i += blockDim.x) {
    float v = 0.0f;
    if (i < 2047) {
      int delta = i - 1023;                 // s - t
      int bucket = (delta > 0) ? 16 : 0;    // nb = 16
      int rp = (delta < 0) ? -delta : delta;
      if (rp < 8) {
        bucket += rp;
      } else {
        // 8 + log(rp/8)/log(16) * 8, clamped to 15
        int rl = 8 + (int)(logf((float)rp * 0.125f) * (8.0f / logf(16.0f)));
        bucket += (rl > 15) ? 15 : rl;
      }
      v = tbl[bucket * Hn + h];
    }
    out[h * 2048 + i] = v;
  }
}

// ---------------------------------------------------------------------------
// bf16 GEMM: out = A (Mn x K, row-major) @ Bw^T (Bw is N x K row-major) + bias
// N fixed at En. 128x128 workgroup tile, 8 waves of 32x64, K-chunk 32.
// mode 0: write bf16 into head-split (b,h,t,d) layout (for Q/K/V)
// mode 1: write fp32 row-major (final output)
// ---------------------------------------------------------------------------
__global__ __launch_bounds__(256)
void gemm_bf16_kernel(const __bf16* __restrict__ A, const __bf16* __restrict__ Bw,
                      const float* __restrict__ bias, void* __restrict__ out,
                      int K, int mode) {
  __shared__ __bf16 Asl[128 * 32];
  __shared__ __bf16 Bsl[128 * 32];

  const int tid  = threadIdx.x;
  const int lane = tid & 31;
  const int wave = tid >> 5;
  const int waveM = wave & 3;   // 0..3 -> 32-row slabs
  const int waveN = wave >> 2;  // 0..1 -> 64-col slabs
  const int m0 = blockIdx.y * 128;
  const int n0 = blockIdx.x * 128;

  const v8f vzero = {0.f,0.f,0.f,0.f,0.f,0.f,0.f,0.f};
  v8f acc[2][4];
#pragma unroll
  for (int i = 0; i < 2; i++)
#pragma unroll
    for (int j = 0; j < 4; j++) acc[i][j] = vzero;

  const int ldrow = tid >> 1;
  const int ldcol = (tid & 1) * 16;

  for (int kc = 0; kc < K; kc += 32) {
    // stage 128x32 bf16 tiles of A and W (both K-contiguous in memory)
    *(uint4*)&Asl[ldrow * 32 + ldcol] =
        *(const uint4*)&A[(size_t)(m0 + ldrow) * K + kc + ldcol];
    *(uint4*)&Bsl[ldrow * 32 + ldcol] =
        *(const uint4*)&Bw[(size_t)(n0 + ldrow) * K + kc + ldcol];
    __syncthreads();

    // A fragment: lane holds row (lane%16), K chunks [g..g+7],[g+16..g+23]
    FragAB af[2], bf[4];
    const int g = (lane >> 4) * 8;
#pragma unroll
    for (int mt = 0; mt < 2; mt++) {
      const int row = waveM * 32 + mt * 16 + (lane & 15);
      af[mt].u[0] = *(const uint4*)&Asl[row * 32 + g];
      af[mt].u[1] = *(const uint4*)&Asl[row * 32 + g + 16];
    }
    // B fragment: lane holds col (lane%16), K chunks [koff..koff+7],[koff+16..koff+23]
    const int koff = (lane >> 4) * 8;
#pragma unroll
    for (int nt = 0; nt < 4; nt++) {
      const int col = waveN * 64 + nt * 16 + (lane & 15);
      bf[nt].u[0] = *(const uint4*)&Bsl[col * 32 + koff];
      bf[nt].u[1] = *(const uint4*)&Bsl[col * 32 + koff + 16];
    }
#pragma unroll
    for (int mt = 0; mt < 2; mt++)
#pragma unroll
      for (int nt = 0; nt < 4; nt++)
        acc[mt][nt] = __builtin_amdgcn_wmma_f32_16x16x32_bf16(
            false, af[mt].v, false, bf[nt].v, (short)0, acc[mt][nt],
            false, false);
    __syncthreads();
  }

  // epilogue: C layout -> VGPR r holds row (r + 8*(lane/16)), col lane%16
#pragma unroll
  for (int mt = 0; mt < 2; mt++)
#pragma unroll
    for (int nt = 0; nt < 4; nt++)
#pragma unroll
      for (int r = 0; r < 8; r++) {
        const int m = m0 + waveM * 32 + mt * 16 + r + ((lane >> 4) << 3);
        const int n = n0 + waveN * 64 + nt * 16 + (lane & 15);
        const float v = acc[mt][nt][r] + bias[n];
        if (mode == 0) {
          const int t = m >> 2, b = m & 3;      // m = t*B + b
          const int h = n >> 6, d = n & 63;     // n = h*64 + d
          ((__bf16*)out)[((((size_t)b * Hn + h) * Tn + t) << 6) + d] = (__bf16)v;
        } else {
          ((float*)out)[(size_t)m * En + n] = v;
        }
      }
}

// ---------------------------------------------------------------------------
// Gate: per (b,h,t) compute sigmoid(sum4) gates from unscaled Q.
// gate = sig(A)*(sig(B)*grep_a[h] - 1) + 2
// ---------------------------------------------------------------------------
__global__ void gate_kernel(const __bf16* __restrict__ Qh,
                            const float* __restrict__ gW,
                            const float* __restrict__ gb,
                            const float* __restrict__ ga,
                            float* __restrict__ gates) {
  const int idx = blockIdx.x * blockDim.x + threadIdx.x;  // bh*T + t
  if (idx >= Bn * Hn * Tn) return;
  const int h = (idx >> 10) & 15;
  const __bf16* q = Qh + ((size_t)idx << 6);
  float qf[64];
#pragma unroll
  for (int d = 0; d < 64; d++) qf[d] = (float)q[d];
  float sa = 0.f, sb = 0.f;
  for (int i = 0; i < 8; i++) {
    float dot = gb[i];
    for (int d = 0; d < 64; d++) dot += qf[d] * gW[i * 64 + d];
    if (i < 4) sa += dot; else sb += dot;
  }
  const float A  = 1.0f / (1.0f + __expf(-sa));
  const float Bv = 1.0f / (1.0f + __expf(-sb));
  gates[idx] = A * (Bv * ga[h] - 1.0f) + 2.0f;
}

// ---------------------------------------------------------------------------
// Attention: one (b,h) per workgroup-column; whole K + V^T staged in LDS
// (256 KB) -- CDNA5's 320 KB WGP LDS makes full-context staging possible.
// K is staged by the Tensor Data Mover (tensor_load_to_lds, TENSORcnt).
// 8 waves x 16 query rows, flash-style online softmax, all GEMMs via WMMA.
// logits = S * 0.125 + gate(t) * bias(s - t)   (ALPHA cancels algebraically)
// ---------------------------------------------------------------------------
__global__ __launch_bounds__(256)
void attn_kernel(const __bf16* __restrict__ Qh, const __bf16* __restrict__ Kh,
                 const __bf16* __restrict__ Vh, const float* __restrict__ gates,
                 const float* __restrict__ bias_tab, __bf16* __restrict__ Ctx) {
  extern __shared__ char smem[];
  __bf16* Klds  = (__bf16*)smem;                   // [1024][64]   128 KB  (LDS off 0)
  __bf16* Vtlds = (__bf16*)(smem + 131072);        // [64][1024]   128 KB
  float*  biasl = (float*)(smem + 262144);         // [2048]         8 KB
  __bf16* Plds  = (__bf16*)(smem + 270336);        // [8][16*32]     8 KB

  const int bh   = blockIdx.y;
  const int b    = bh >> 4, h = bh & 15;
  const int tid  = threadIdx.x;
  const int lane = tid & 31;
  const int wave = tid >> 5;
  const int t0   = blockIdx.x * 128 + wave * 16;
  const size_t base = (size_t)bh * Tn * Dn;

  // ---- K stage: one TDM descriptor moves the whole 128 KB K block ----
  // D# per CDNA5 ISA ch.8: 1-D tile of 16384 x 8-byte units, no cluster
  // multicast, no padding/iteration. Issued by wave 0 only; completion is
  // tracked with TENSORcnt and published to the workgroup via the barrier.
  if (wave == 0) {
    const unsigned long long ga = (unsigned long long)(Kh + base); // byte addr
    const unsigned lds_dst = 0;            // Klds == dynamic-LDS base
    u32x4 g0;
    g0[0] = 1u;                            // count=1, is_restore=0, gather=0
    g0[1] = lds_dst;                       // [63:32]  lds_addr
    g0[2] = (unsigned)(ga & 0xffffffffu);  // [95:64]  global_addr lo
    g0[3] = (unsigned)((ga >> 32) & 0x01ffffffu) | (2u << 30); // addr hi | type=2

    const unsigned long long nunits = (Tn * Dn * 2) / 8;  // 16384 x 8B
    unsigned long long q0 = (3ull << 16)                  // data_size = 8B
                          | ((nunits & 0xffffull) << 48); // tensor_dim0 lo
    unsigned long long q1 = ((nunits >> 16) & 0xffffull)  // tensor_dim0 hi
                          | (1ull << 16)                  // tensor_dim1 = 1
                          | ((nunits & 0xffffull) << 48); // tile_dim0
    unsigned long long q2 = (nunits << 32);               // tensor_dim0_stride lo
    unsigned long long q3 = (nunits >> 32) & 0xffffull;   // stride hi
    u32x8 g1;
    g1[0] = (unsigned)q0;  g1[1] = (unsigned)(q0 >> 32);
    g1[2] = (unsigned)q1;  g1[3] = (unsigned)(q1 >> 32);
    g1[4] = (unsigned)q2;  g1[5] = (unsigned)(q2 >> 32);
    g1[6] = (unsigned)q3;  g1[7] = (unsigned)(q3 >> 32);

    asm volatile("tensor_load_to_lds %0, %1" : : "s"(g0), "s"(g1) : "memory");
  }

  // ---- V stage: transposed (d-major, s contiguous) for PV B-fragments ----
  for (int i = tid; i < (Tn * Dn) / 8; i += 256) {
    const int s = i >> 3, d0 = (i & 7) * 8;
    uint4 u = *(const uint4*)&Vh[base + (size_t)i * 8];
    const __bf16* e = (const __bf16*)&u;
#pragma unroll
    for (int j = 0; j < 8; j++) Vtlds[(d0 + j) * Tn + s] = e[j];
  }
  for (int i = tid; i < 2048; i += 256) biasl[i] = bias_tab[h * 2048 + i];

  if (wave == 0) __builtin_amdgcn_s_wait_tensorcnt(0);  // K DMA done
  __syncthreads();

  // Q fragments for this wave's 16 rows (K = 64 -> two 16x32 chunks)
  FragAB af[2];
  {
    const int t = t0 + (lane & 15);
    const __bf16* q = Qh + base + (size_t)t * 64;
    const int g = (lane >> 4) * 8;
#pragma unroll
    for (int kk = 0; kk < 2; kk++) {
      af[kk].u[0] = *(const uint4*)&q[kk * 32 + g];
      af[kk].u[1] = *(const uint4*)&q[kk * 32 + g + 16];
    }
  }
  float gr[8];
  int   trow[8];
#pragma unroll
  for (int r = 0; r < 8; r++) {
    trow[r] = t0 + r + ((lane >> 4) << 3);
    gr[r]   = gates[bh * Tn + trow[r]];
  }

  const v8f vzero = {0.f,0.f,0.f,0.f,0.f,0.f,0.f,0.f};
  v8f Ot[4];
#pragma unroll
  for (int nt = 0; nt < 4; nt++) Ot[nt] = vzero;
  float mrun[8], lrun[8];
#pragma unroll
  for (int r = 0; r < 8; r++) { mrun[r] = -3.0e38f; lrun[r] = 0.f; }

  const int koffB = (lane >> 4) * 8;
  for (int s0 = 0; s0 < Tn; s0 += 32) {
    // ---- S = Q @ K^T for a 16x32 score slab (two 16x16 C tiles) ----
    v8f c0 = vzero, c1 = vzero;
#pragma unroll
    for (int kk = 0; kk < 2; kk++) {
      FragAB bk0, bk1;
      const int s_a = s0 + (lane & 15);
      const int s_b = s_a + 16;
      bk0.u[0] = *(const uint4*)&Klds[s_a * 64 + kk * 32 + koffB];
      bk0.u[1] = *(const uint4*)&Klds[s_a * 64 + kk * 32 + koffB + 16];
      bk1.u[0] = *(const uint4*)&Klds[s_b * 64 + kk * 32 + koffB];
      bk1.u[1] = *(const uint4*)&Klds[s_b * 64 + kk * 32 + koffB + 16];
      c0 = __builtin_amdgcn_wmma_f32_16x16x32_bf16(false, af[kk].v, false,
                                                   bk0.v, (short)0, c0, false, false);
      c1 = __builtin_amdgcn_wmma_f32_16x16x32_bf16(false, af[kk].v, false,
                                                   bk1.v, (short)0, c1, false, false);
    }

    // ---- logits + online softmax, P staged to LDS as bf16 A-fragment ----
    __bf16* P = Plds + wave * 512;
#pragma unroll
    for (int r = 0; r < 8; r++) {
      const int sc0 = s0 + (lane & 15);
      const float l0 = c0[r] * 0.125f + gr[r] * biasl[sc0 - trow[r] + 1023];
      const float l1 = c1[r] * 0.125f + gr[r] * biasl[sc0 + 16 - trow[r] + 1023];
      float v = fmaxf(l0, l1);
#pragma unroll
      for (int off = 8; off >= 1; off >>= 1) v = fmaxf(v, __shfl_xor(v, off, 32));
      const float mn    = fmaxf(mrun[r], v);
      const float scale = __expf(mrun[r] - mn);
      const float p0 = __expf(l0 - mn);
      const float p1 = __expf(l1 - mn);
      float rs = p0 + p1;
#pragma unroll
      for (int off = 8; off >= 1; off >>= 1) rs += __shfl_xor(rs, off, 32);
      lrun[r] = lrun[r] * scale + rs;
      mrun[r] = mn;
#pragma unroll
      for (int nt = 0; nt < 4; nt++) Ot[nt][r] = Ot[nt][r] * scale;
      const int prow = r + ((lane >> 4) << 3);
      P[prow * 32 + (lane & 15)]      = (__bf16)p0;
      P[prow * 32 + 16 + (lane & 15)] = (__bf16)p1;
    }

    // ---- O += P @ V (V^T staged -> B-fragment is contiguous-K loads) ----
    FragAB pa;
    {
      const int g = (lane >> 4) * 8;
      pa.u[0] = *(const uint4*)&P[(lane & 15) * 32 + g];
      pa.u[1] = *(const uint4*)&P[(lane & 15) * 32 + g + 16];
    }
#pragma unroll
    for (int nt = 0; nt < 4; nt++) {
      FragAB bv;
      const int d  = nt * 16 + (lane & 15);
      const int so = s0 + koffB;
      bv.u[0] = *(const uint4*)&Vtlds[d * Tn + so];
      bv.u[1] = *(const uint4*)&Vtlds[d * Tn + so + 16];
      Ot[nt] = __builtin_amdgcn_wmma_f32_16x16x32_bf16(false, pa.v, false,
                                                       bv.v, (short)0, Ot[nt],
                                                       false, false);
    }
  }

  // ---- finalize: divide by l, write context in (t*B+b, h*64+d) bf16 ----
#pragma unroll
  for (int r = 0; r < 8; r++) {
    const float inv = 1.0f / lrun[r];
    const int m = trow[r] * Bn + b;
#pragma unroll
    for (int nt = 0; nt < 4; nt++) {
      const int col = h * 64 + nt * 16 + (lane & 15);
      Ctx[(size_t)m * En + col] = (__bf16)(Ot[nt][r] * inv);
    }
  }
}

// ---------------------------------------------------------------------------
// Host-side launch
// ---------------------------------------------------------------------------
extern "C" void kernel_launch(void* const* d_in, const int* in_sizes, int n_in,
                              void* d_out, int out_size, void* d_ws, size_t ws_size,
                              hipStream_t stream) {
  const float* query = (const float*)d_in[0];
  const float* Wq = (const float*)d_in[1];
  const float* bq = (const float*)d_in[2];
  const float* Wk = (const float*)d_in[3];
  const float* bk = (const float*)d_in[4];
  const float* Wv = (const float*)d_in[5];
  const float* bv = (const float*)d_in[6];
  const float* Wo = (const float*)d_in[7];
  const float* bo = (const float*)d_in[8];
  const float* relb = (const float*)d_in[9];
  const float* gW = (const float*)d_in[10];
  const float* gb = (const float*)d_in[11];
  const float* ga = (const float*)d_in[12];

  char* ws = (char*)d_ws;
  size_t off = 0;
  auto alloc = [&](size_t bytes) -> char* {
    char* p = ws + off;
    off += (bytes + 255) & ~(size_t)255;
    return p;
  };
  __bf16* Xbf  = (__bf16*)alloc((size_t)Mn * En * 2);
  __bf16* Wqb  = (__bf16*)alloc((size_t)En * En * 2);
  __bf16* Wkb  = (__bf16*)alloc((size_t)En * En * 2);
  __bf16* Wvb  = (__bf16*)alloc((size_t)En * En * 2);
  __bf16* Wob  = (__bf16*)alloc((size_t)En * En * 2);
  __bf16* Qh   = (__bf16*)alloc((size_t)Bn * Hn * Tn * Dn * 2);
  __bf16* Khp  = (__bf16*)alloc((size_t)Bn * Hn * Tn * Dn * 2);
  __bf16* Vhp  = (__bf16*)alloc((size_t)Bn * Hn * Tn * Dn * 2);
  float*  gates= (float*)alloc((size_t)Bn * Hn * Tn * 4);
  float*  btab = (float*)alloc((size_t)Hn * 2048 * 4);
  __bf16* Ctx  = (__bf16*)alloc((size_t)Mn * En * 2);

  cvt_bf16_kernel<<<512, 256, 0, stream>>>(query, Xbf, Mn * En);
  cvt_bf16_kernel<<<512, 256, 0, stream>>>(Wq, Wqb, En * En);
  cvt_bf16_kernel<<<512, 256, 0, stream>>>(Wk, Wkb, En * En);
  cvt_bf16_kernel<<<512, 256, 0, stream>>>(Wv, Wvb, En * En);
  cvt_bf16_kernel<<<512, 256, 0, stream>>>(Wo, Wob, En * En);
  bias_table_kernel<<<Hn, 256, 0, stream>>>(relb, btab);

  dim3 gg(En / 128, Mn / 128);
  gemm_bf16_kernel<<<gg, 256, 0, stream>>>(Xbf, Wqb, bq, Qh,  En, 0);
  gemm_bf16_kernel<<<gg, 256, 0, stream>>>(Xbf, Wkb, bk, Khp, En, 0);
  gemm_bf16_kernel<<<gg, 256, 0, stream>>>(Xbf, Wvb, bv, Vhp, En, 0);

  gate_kernel<<<(Bn * Hn * Tn) / 256, 256, 0, stream>>>(Qh, gW, gb, ga, gates);

  const size_t smem_bytes = 278528;  // 272 KB of the 320 KB WGP LDS
  attn_kernel<<<dim3(Tn / 128, Bn * Hn), 256, smem_bytes, stream>>>(
      Qh, Khp, Vhp, gates, btab, Ctx);

  gemm_bf16_kernel<<<gg, 256, 0, stream>>>(Ctx, Wob, bo, d_out, En, 1);
}
